// NonLocalOp_10118942949559
// MI455X (gfx1250) — matrix operand order repaired and checked by reading, not compiled, not run on hardware
//
#include <hip/hip_runtime.h>
#include <stdint.h>

#define B_  8
#define C_  256
#define HW_ 4096
#define E_  128

typedef unsigned short u16;
typedef __attribute__((ext_vector_type(16))) __bf16        v16bf;
typedef __attribute__((ext_vector_type(8)))  float         v8f;
typedef __attribute__((ext_vector_type(4)))  unsigned int  u32x4;
typedef __attribute__((ext_vector_type(8)))  int           i32x8;
typedef __attribute__((ext_vector_type(4)))  int           i32x4;

#ifndef __has_builtin
#define __has_builtin(x) 0
#endif
#if __has_builtin(__builtin_amdgcn_tensor_load_to_lds) && __has_builtin(__builtin_amdgcn_s_wait_tensorcnt)
#define HAVE_TDM 1
#else
#define HAVE_TDM 0
#endif

union ABf {
  u32x4 u4[2];
  unsigned int u[8];
  unsigned long long d[4];
  v16bf v;
};

__device__ __forceinline__ u16 f2bf(float x) {
  unsigned u = __float_as_uint(x);
  u += 0x7fffu + ((u >> 16) & 1u);  // round-to-nearest-even-ish
  return (u16)(u >> 16);
}

__device__ __forceinline__ v8f wmma_bf16(v16bf a, v16bf b, v8f c) {
  return __builtin_amdgcn_wmma_f32_16x16x32_bf16(false, a, false, b, (short)0, c, false, false);
}

#if HAVE_TDM
// 2D TDM load: tile (tile1 rows x tile0 elems) of bf16 tensor (row stride stride0 elems)
// into LDS with hardware padding (pad_amt+1 dwords after every interval) for bank/alignment.
__device__ __forceinline__ void tdm_load_2d(unsigned lds_addr, const void* gp,
                                            unsigned tdim0, unsigned tdim1,
                                            unsigned tile0, unsigned tile1,
                                            unsigned stride0, unsigned pad_int,
                                            unsigned pad_amt) {
  unsigned long long ga = (unsigned long long)(uintptr_t)gp;
  u32x4 g0;
  g0[0] = 1u;                                   // count=1, user desc
  g0[1] = lds_addr;                             // LDS byte address
  g0[2] = (unsigned)ga;                         // global addr lo
  g0[3] = (unsigned)((ga >> 32) & 0x1ffffffull) | (2u << 30);  // addr hi + type=2
  i32x8 g1;
  g1[0] = (int)((1u << 16)                      // data_size = 2B
              | (1u << 20)                      // pad_enable
              | (pad_int << 22) | (pad_amt << 25));
  g1[1] = (int)((tdim0 & 0xffffu) << 16);                       // tensor_dim0 lo
  g1[2] = (int)((tdim0 >> 16) | ((tdim1 & 0xffffu) << 16));     // dim0 hi | dim1 lo
  g1[3] = (int)((tdim1 >> 16) | (tile0 << 16));                 // dim1 hi | tile_dim0
  g1[4] = (int)tile1;                                           // tile_dim1 (tile_dim2=0)
  g1[5] = (int)stride0;                                         // tensor_dim0_stride lo
  g1[6] = 0; g1[7] = 0;
  i32x4 z4 = {0, 0, 0, 0};
#if defined(__clang_major__) && (__clang_major__ >= 23)
  i32x8 z8 = {0, 0, 0, 0, 0, 0, 0, 0};
  __builtin_amdgcn_tensor_load_to_lds(g0, g1, z4, z4, z8, 0);
#else
  __builtin_amdgcn_tensor_load_to_lds(g0, g1, z4, z4, 0);
#endif
}
#endif

// ---------------------------------------------------------------------------
// Kernel 1: l [C,HW] f32 -> gb bf16 [C,HW] and lT bf16 [HW,C] (LDS transpose)
// ---------------------------------------------------------------------------
__global__ __launch_bounds__(256) void convt_kernel(const float* __restrict__ l,
                                                    u16* __restrict__ gb,
                                                    u16* __restrict__ lT) {
  __shared__ u16 T[32][33];
  const int tx = threadIdx.x, ty = threadIdx.y;
  const int q0 = blockIdx.x * 32, c0 = blockIdx.y * 32, b = blockIdx.z;
  const float* lb = l + (size_t)b * C_ * HW_;
  u16* gbB = gb + (size_t)b * C_ * HW_;
  u16* lTB = lT + (size_t)b * HW_ * C_;
#pragma unroll
  for (int k = 0; k < 4; ++k) {
    int c = c0 + ty + 8 * k;
    u16 hv = f2bf(lb[(size_t)c * HW_ + q0 + tx]);
    gbB[(size_t)c * HW_ + q0 + tx] = hv;
    T[ty + 8 * k][tx] = hv;
  }
  __syncthreads();
#pragma unroll
  for (int k = 0; k < 4; ++k) {
    int qr = ty + 8 * k;
    lTB[(size_t)(q0 + qr) * C_ + c0 + tx] = T[tx][qr];
  }
}

// ---------------------------------------------------------------------------
// Kernel 2: weights f32 [E,C] -> bf16
// ---------------------------------------------------------------------------
__global__ __launch_bounds__(256) void convw_kernel(const float* __restrict__ wt,
                                                    const float* __restrict__ wp,
                                                    u16* __restrict__ wtb,
                                                    u16* __restrict__ wpb) {
  int i = blockIdx.x * 256 + threadIdx.x;
  if (i < E_ * C_) wtb[i] = f2bf(wt[i]);
  else             wpb[i - E_ * C_] = f2bf(wp[i - E_ * C_]);
}

// ---------------------------------------------------------------------------
// Kernel 3: theta/phi = lT @ w^T  via WMMA bf16.
// Each wave: 16 q-rows x (128 theta + 128 phi) cols. Block = 8 waves = 128 rows.
// ---------------------------------------------------------------------------
__global__ __launch_bounds__(256) void proj_kernel(const u16* __restrict__ lT,
                                                   const u16* __restrict__ wtb,
                                                   const u16* __restrict__ wpb,
                                                   u16* __restrict__ theta,
                                                   u16* __restrict__ phi) {
  __shared__ __align__(16) u16 Tb[8][16 * 136];  // per-wave transpose buffer
  const int tid = threadIdx.x;
  const int wave = tid >> 5, lane = tid & 31;
  const int h = lane >> 4, l16 = lane & 15;
  const int b = blockIdx.y;
  const int q0w = blockIdx.x * 128 + wave * 16;
  const u16* aB = lT + (size_t)b * HW_ * C_ + (size_t)(q0w + l16) * C_;

  v8f acc[16];
#pragma unroll
  for (int t = 0; t < 16; ++t)
#pragma unroll
    for (int r = 0; r < 8; ++r) acc[t][r] = 0.0f;

#pragma unroll
  for (int ck = 0; ck < 8; ++ck) {
    ABf a;
    const u16* ap = aB + ck * 32 + 16 * h;       // A: K 0-15 (h0), 16-31 (h1)
    a.u4[0] = *(const u32x4*)ap;
    a.u4[1] = *(const u32x4*)(ap + 8);
#pragma unroll
    for (int t = 0; t < 16; ++t) {
      const u16* wsel = (t < 8) ? wtb : wpb;
      const u16* wp2 = wsel + (size_t)(((t & 7) * 16) + l16) * C_ + ck * 32 + 16 * h;
      ABf bf;                                     // B: lane=col e, K 0-15 (h0) / 16-31 (h1)
      bf.u4[0] = *(const u32x4*)wp2;
      bf.u4[1] = *(const u32x4*)(wp2 + 8);
      acc[t] = wmma_bf16(a.v, bf.v, acc[t]);
    }
  }
  // transpose through LDS for coalesced [q][e] stores
#pragma unroll
  for (int sel = 0; sel < 2; ++sel) {
#pragma unroll
    for (int t = 0; t < 8; ++t)
#pragma unroll
      for (int r = 0; r < 8; ++r)
        Tb[wave][(r + 8 * h) * 136 + t * 16 + l16] = f2bf(acc[sel * 8 + t][r]);
    u16* dst = (sel ? phi : theta) + (size_t)b * HW_ * E_;
#pragma unroll
    for (int i = 0; i < 4; ++i) {
      int idx = i * 32 + lane;
      int rw = idx >> 3, chv = idx & 7;
      u32x4 vv = *(const u32x4*)&Tb[wave][rw * 136 + chv * 16];
      *(u32x4*)(dst + (size_t)(q0w + rw) * E_ + chv * 16) = vv;
    }
  }
}

// ---------------------------------------------------------------------------
// Kernel 4: flash attention. Block = 128 queries (8 waves x 16). K/V tiles
// TDM-staged into LDS, double buffered. Per 32-key step: 8 WMMA (QK^T),
// online softmax, P via LDS transpose, 16 WMMA (PV) into full C=256 accum.
// ---------------------------------------------------------------------------
__global__ __launch_bounds__(256) void attn_kernel(const u16* __restrict__ theta,
                                                   const u16* __restrict__ phi,
                                                   const u16* __restrict__ gb,
                                                   float* __restrict__ out) {
  __shared__ __align__(16) u16 Kt[2][32 * 136];   // 32 keys x 128 E, pitch 136 (TDM pad 4dw/64dw)
  __shared__ __align__(16) u16 Vt[2][256 * 36];   // 256 ch x 32 keys, pitch 36 (TDM pad 2dw/16dw)
  __shared__ __align__(16) u16 Pb[8][16 * 34];    // per-wave P / output transpose buffer

  const int tid = threadIdx.x;
  const int wave = tid >> 5, lane = tid & 31;
  const int h = lane >> 4, l16 = lane & 15;
  const int b = blockIdx.y;
  const int q0w = blockIdx.x * 128 + wave * 16;

  const u16* phiB = phi + (size_t)b * HW_ * E_;
  const u16* gbB = gb + (size_t)b * C_ * HW_;

  // Q fragments (resident all loop)
  ABf qf[4];
  {
    const u16* qp = theta + (size_t)b * HW_ * E_ + (size_t)(q0w + l16) * E_;
#pragma unroll
    for (int c = 0; c < 4; ++c) {
      qf[c].u4[0] = *(const u32x4*)(qp + c * 32 + 16 * h);
      qf[c].u4[1] = *(const u32x4*)(qp + c * 32 + 16 * h + 8);
    }
  }

  v8f o[16];
#pragma unroll
  for (int t = 0; t < 16; ++t)
#pragma unroll
    for (int r = 0; r < 8; ++r) o[t][r] = 0.0f;
  float m[8], lsum[8];
#pragma unroll
  for (int r = 0; r < 8; ++r) { m[r] = -1e30f; lsum[r] = 0.0f; }

  const int nIter = HW_ / 32;

#if HAVE_TDM
  const bool isW0 = (__builtin_amdgcn_readfirstlane(wave) == 0);
  if (isW0) {
    tdm_load_2d((unsigned)(uintptr_t)&Kt[0][0], phiB, E_, HW_, E_, 32, E_, 5, 3);
    tdm_load_2d((unsigned)(uintptr_t)&Vt[0][0], gbB, HW_, C_, 32, C_, HW_, 3, 1);
  }
#endif

  for (int it = 0; it < nIter; ++it) {
    const int k0 = it * 32;
#if HAVE_TDM
    const int cur = it & 1;
    if (isW0) {
      if (it + 1 < nIter) {
        const int nb = (it + 1) & 1;
        tdm_load_2d((unsigned)(uintptr_t)&Kt[nb][0], phiB + (size_t)(k0 + 32) * E_,
                    E_, HW_, E_, 32, E_, 5, 3);
        tdm_load_2d((unsigned)(uintptr_t)&Vt[nb][0], gbB + (k0 + 32),
                    HW_, C_, 32, C_, HW_, 3, 1);
        __builtin_amdgcn_s_wait_tensorcnt(2);  // current pair done, next in flight
      } else {
        __builtin_amdgcn_s_wait_tensorcnt(0);
      }
    }
#else
    const int cur = 0;
    for (int i = tid; i < 32 * E_; i += 256) {
      int rk = i >> 7, ce = i & 127;
      Kt[0][rk * 136 + ce] = phiB[(size_t)(k0 + rk) * E_ + ce];
    }
    for (int i = tid; i < C_ * 32; i += 256) {
      int rc = i >> 5, ck2 = i & 31;
      Vt[0][rc * 36 + ck2] = gbB[(size_t)rc * HW_ + k0 + ck2];
    }
#endif
    __syncthreads();

    // ---- S = Q K^T (two 16-key tiles) ----
    v8f s0, s1;
#pragma unroll
    for (int r = 0; r < 8; ++r) { s0[r] = 0.0f; s1[r] = 0.0f; }
    const u16* Kb = &Kt[cur][0];
#pragma unroll
    for (int c = 0; c < 4; ++c) {
      ABf kf0, kf1;
      const u16* kp0 = Kb + l16 * 136 + c * 32 + 16 * h;
      const u16* kp1 = kp0 + 16 * 136;
      kf0.u4[0] = *(const u32x4*)kp0;
      kf0.u4[1] = *(const u32x4*)(kp0 + 8);
      kf1.u4[0] = *(const u32x4*)kp1;
      kf1.u4[1] = *(const u32x4*)(kp1 + 8);
      s0 = wmma_bf16(qf[c].v, kf0.v, s0);
      s1 = wmma_bf16(qf[c].v, kf1.v, s1);
    }

    // ---- online softmax (base-2 domain), rows spread over 16-lane groups ----
    const float SC = 0.127478859f;  // (1/sqrt(128)) * log2(e)
    float alpha[8];
#pragma unroll
    for (int r = 0; r < 8; ++r) {
      float a0 = s0[r] * SC, a1 = s1[r] * SC;
      float mx = fmaxf(a0, a1);
      mx = fmaxf(mx, __shfl_xor(mx, 1));
      mx = fmaxf(mx, __shfl_xor(mx, 2));
      mx = fmaxf(mx, __shfl_xor(mx, 4));
      mx = fmaxf(mx, __shfl_xor(mx, 8));
      float mn = fmaxf(m[r], mx);
      float al = exp2f(m[r] - mn);
      m[r] = mn;
      float p0 = exp2f(a0 - mn);
      float p1 = exp2f(a1 - mn);
      float rs = p0 + p1;
      rs += __shfl_xor(rs, 1);
      rs += __shfl_xor(rs, 2);
      rs += __shfl_xor(rs, 4);
      rs += __shfl_xor(rs, 8);
      lsum[r] = lsum[r] * al + rs;
      alpha[r] = al;
      s0[r] = p0; s1[r] = p1;
    }
#pragma unroll
    for (int t = 0; t < 16; ++t)
#pragma unroll
      for (int r = 0; r < 8; ++r) o[t][r] *= alpha[r];

    // ---- P (C-layout) -> LDS -> A-fragment layout (wave-private, in-order DS) ----
    u16* pw = &Pb[wave][0];
#pragma unroll
    for (int r = 0; r < 8; ++r) {
      int rq = r + 8 * h;
      pw[rq * 34 + l16] = f2bf(s0[r]);
      pw[rq * 34 + 16 + l16] = f2bf(s1[r]);
    }
    ABf pf;
    {
      const u16* pr = pw + l16 * 34 + 16 * h;
#pragma unroll
      for (int d2 = 0; d2 < 4; ++d2) {
        pf.u[d2]     = *(const unsigned int*)(pr + 2 * d2);
        pf.u[4 + d2] = *(const unsigned int*)(pr + 8 + 2 * d2);
      }
    }

    // ---- O += P V over all 256 channels ----
    const u16* Vb = &Vt[cur][0];
#pragma unroll
    for (int t = 0; t < 16; ++t) {
      ABf vf;
      const unsigned long long* vp =
          (const unsigned long long*)(Vb + (t * 16 + l16) * 36 + 16 * h);
      vf.d[0] = vp[0]; vf.d[1] = vp[1]; vf.d[2] = vp[2]; vf.d[3] = vp[3];
      o[t] = wmma_bf16(pf.v, vf.v, o[t]);
    }
    __syncthreads();
  }

  // ---- epilogue: O / lsum, transpose via LDS, coalesced [c][q] stores ----
  float rl[8];
#pragma unroll
  for (int r = 0; r < 8; ++r) rl[r] = 1.0f / lsum[r];
  float* Pf = (float*)&Pb[wave][0];  // 16x17 f32 scratch
  float* outB = out + (size_t)b * C_ * HW_;
#pragma unroll
  for (int t = 0; t < 16; ++t) {
#pragma unroll
    for (int r = 0; r < 8; ++r)
      Pf[l16 * 17 + 8 * h + r] = o[t][r] * rl[r];
#pragma unroll
    for (int i = 0; i < 8; ++i) {
      int rc = i * 2 + h;
      outB[(size_t)(t * 16 + rc) * HW_ + q0w + l16] = Pf[rc * 17 + l16];
    }
  }
}

// ---------------------------------------------------------------------------
extern "C" void kernel_launch(void* const* d_in, const int* in_sizes, int n_in,
                              void* d_out, int out_size, void* d_ws, size_t ws_size,
                              hipStream_t stream) {
  (void)in_sizes; (void)n_in; (void)out_size; (void)ws_size;
  const float* l = (const float*)d_in[0];
  const float* w_theta = (const float*)d_in[1];
  const float* w_phi = (const float*)d_in[2];
  float* out = (float*)d_out;

  char* ws = (char*)d_ws;
  const size_t szGB = (size_t)B_ * C_ * HW_ * 2;  // 16 MB
  const size_t szTH = (size_t)B_ * HW_ * E_ * 2;  // 8 MB
  u16* gb  = (u16*)(ws);
  u16* lT  = (u16*)(ws + szGB);
  u16* th  = (u16*)(ws + 2 * szGB);
  u16* ph  = (u16*)(ws + 2 * szGB + szTH);
  u16* wtb = (u16*)(ws + 2 * szGB + 2 * szTH);
  u16* wpb = (u16*)(ws + 2 * szGB + 2 * szTH + (size_t)E_ * C_ * 2);

  convt_kernel<<<dim3(HW_ / 32, C_ / 32, B_), dim3(32, 8), 0, stream>>>(l, gb, lT);
  convw_kernel<<<dim3((2 * E_ * C_) / 256), dim3(256), 0, stream>>>(w_theta, w_phi, wtb, wpb);
  proj_kernel<<<dim3(HW_ / 128, B_), dim3(256), 0, stream>>>(lT, wtb, wpb, th, ph);
  attn_kernel<<<dim3(HW_ / 128, B_), dim3(256), 0, stream>>>(th, ph, gb, out);
}
